// HeteroGNN_12970801234251
// MI455X (gfx1250) — matrix operand order repaired and checked by reading, not compiled, not run on hardware
//
#include <hip/hip_runtime.h>
#include <hip/hip_bf16.h>
#include <stdint.h>

// ---------------------------------------------------------------------------
// HeteroGNN (2-layer hetero SAGE, H=128) for MI455X / gfx1250, wave32 + WMMA.
//   Phase 1 (per relation): wave-per-edge gather of bf16 source rows (b64),
//            f32 global_atomic_add into L2-resident accumulators + degree cnt.
//   Phase 2: fused WMMA GEMM. A operands (mean-scaled acc / x_dst) are staged
//            once per block into LDS (bank-conflict-free padded rows) and
//            shared by all waves; B comes from pre-swizzled bf16 fragments
//            (global_load_b128). Epilogue fuses bias + leaky-ReLU + bf16 store.
// Activations bf16 (halves gather traffic), accumulators & WMMA accum f32.
// ---------------------------------------------------------------------------

#define NA_N 100000
#define NP_N 200000
#define E_N  500000
#define H_N  128
#define OUT_N 64
#define L_N  2
#define LDS_STRIDE 136   // 128 + 8 bf16 pad: 272B rows -> conflict-free b128 reads

typedef __attribute__((ext_vector_type(16))) __bf16 v16bf;
typedef __attribute__((ext_vector_type(8)))  __bf16 v8bf;
typedef __attribute__((ext_vector_type(4)))  __bf16 v4bf;
typedef __attribute__((ext_vector_type(8)))  float  v8f;
typedef __attribute__((ext_vector_type(4)))  float  v4f;

// ---------------- elementwise cast ----------------

__global__ void cast_bf16_kernel(const float* __restrict__ in,
                                 __bf16* __restrict__ out, long n) {
  long i = (long)blockIdx.x * blockDim.x + threadIdx.x;
  if (i < n) out[i] = (__bf16)in[i];
}

// ---------------- weight pack: f32 row-major [128][N] (+optional add) ->
// bf16 per-lane WMMA B fragments.
// Wp[(((kt*(N/16)+nt)*32+lane)*16)+j] = W[(kt*32 + (lane>>4)*8 + (j&7) + (j>>3)*16)*N + nt*16 + (lane&15)]
__global__ void pack_b_f32_kernel(const float* __restrict__ a,
                                  const float* __restrict__ b,
                                  __bf16* __restrict__ Wp, int N) {
  int idx = blockIdx.x * blockDim.x + threadIdx.x;
  int ntiles = N >> 4;
  int total = 4 * ntiles * 32 * 16;
  if (idx >= total) return;
  int j    = idx & 15;
  int lane = (idx >> 4) & 31;
  int nt   = (idx >> 9) % ntiles;
  int kt   = (idx >> 9) / ntiles;
  int k = kt * 32 + ((lane >> 4) << 3) + (j & 7) + ((j >> 3) << 4);
  int n = nt * 16 + (lane & 15);
  float v = a[(long)k * N + n];
  if (b) v += b[(long)k * N + n];
  Wp[idx] = (__bf16)v;
}

// ---------------- scatter: one wave32 per edge, 4 bf16 (b64) / lane ----------------

__global__ void scatter_accum_kernel(const __bf16* __restrict__ xsrc,
                                     const int* __restrict__ src,
                                     const int* __restrict__ dst,
                                     float* __restrict__ acc,
                                     float* __restrict__ cnt, int E) {
  const int lane = threadIdx.x & 31;
  const int wavesPerBlock = blockDim.x >> 5;
  const int waveId = blockIdx.x * wavesPerBlock + (threadIdx.x >> 5);
  const int waveStride = gridDim.x * wavesPerBlock;

  for (int e = waveId; e < E; e += waveStride) {
    // prefetch next iteration's source row (global_prefetch_b8)
    int en = e + waveStride;
    if (en < E) {
      int sn = src[en];
      __builtin_prefetch(xsrc + (long)sn * H_N + lane * 4, 0, 3);
    }
    int s = src[e];
    int d = dst[e];
    v4bf sv = *(const v4bf*)(xsrc + (long)s * H_N + lane * 4);  // 8B aligned
    float* dp = acc + (long)d * H_N + lane * 4;
#pragma unroll
    for (int j = 0; j < 4; ++j) {
      __hip_atomic_fetch_add(dp + j, (float)sv[j], __ATOMIC_RELAXED,
                             __HIP_MEMORY_SCOPE_AGENT);
    }
    if (lane == 0)
      __hip_atomic_fetch_add(cnt + d, 1.0f, __ATOMIC_RELAXED,
                             __HIP_MEMORY_SCOPE_AGENT);
  }
}

// ---------------- WMMA fragment loaders ----------------

// A (16x32 bf16) from LDS-staged strip (row-major, ld = LDS_STRIDE):
// lane holds row lane&15; elements 0..7 = K(k0+koff+0..7),
// 8..15 = K(k0+koff+16..23); koff = 8*(lane>>4). Two ds_load_b128 per frag.
__device__ inline v16bf load_a_lds(const __bf16* __restrict__ s,
                                   int k0, int lane) {
  int row  = lane & 15;
  int koff = (lane >> 4) * 8;
  const __bf16* p = s + row * LDS_STRIDE + k0 + koff;     // 16B aligned
  v8bf lo = *(const v8bf*)p;
  v8bf hi = *(const v8bf*)(p + 16);
  return __builtin_shufflevector(lo, hi, 0, 1, 2, 3, 4, 5, 6, 7,
                                         8, 9, 10, 11, 12, 13, 14, 15);
}

// B from pre-packed fragments: two global b128 loads, wave reads 1KB contiguous.
__device__ inline v16bf load_b_packed(const __bf16* __restrict__ Wp,
                                      int kt, int nt, int ntiles, int lane) {
  const __bf16* p = Wp + ((((long)kt * ntiles + nt) * 32 + lane) << 4);
  v8bf lo = *(const v8bf*)p;
  v8bf hi = *(const v8bf*)(p + 8);
  return __builtin_shufflevector(lo, hi, 0, 1, 2, 3, 4, 5, 6, 7,
                                         8, 9, 10, 11, 12, 13, 14, 15);
}

// ---------------- fused SAGE GEMM ----------------
// out[M,N] = sum_{t<NACC} (A_t / max(cnt_t,1)) @ W_t  (+ Xd @ Wx) + bias(_a+_b)
// block = (N/16) waves; each wave owns one 16x16 output tile of a 16-row strip.
// A terms staged in LDS once per block; K = 128 -> 4 WMMA per term.
template <int NACC, bool HAS_X, bool RELU, bool OUT_BF16>
__global__ void sage_gemm_kernel(const float* __restrict__ A0,
                                 const float* __restrict__ c0,
                                 const float* __restrict__ A1,
                                 const float* __restrict__ c1,
                                 const __bf16* __restrict__ Xd,
                                 const __bf16* __restrict__ W0p,
                                 const __bf16* __restrict__ W1p,
                                 const __bf16* __restrict__ Wxp,
                                 const float* __restrict__ bias_a,
                                 const float* __restrict__ bias_b,
                                 void* __restrict__ out, int N) {
  constexpr int NSRC = NACC + (HAS_X ? 1 : 0);
  __shared__ __bf16 sA[NSRC * 16 * LDS_STRIDE];

  const int lane   = threadIdx.x & 31;
  const int nt     = threadIdx.x >> 5;      // wave id = column tile
  const int ntiles = N >> 4;
  const int row0   = blockIdx.x << 4;

  // ---- stage A operands into LDS (each vec8 group: 16B of one row) ----
  for (int g = threadIdx.x; g < 256; g += blockDim.x) {
    int row = g >> 4;            // 0..15
    int col = (g & 15) * 8;      // 0..120
    int slot = 0;
    if constexpr (NACC >= 1) {
      float sc = 1.0f / fmaxf(c0[row0 + row], 1.0f);
      const float* p = A0 + (long)(row0 + row) * H_N + col;
      v4f f0 = *(const v4f*)p;
      v4f f1 = *(const v4f*)(p + 4);
      v8bf o;
#pragma unroll
      for (int i = 0; i < 4; ++i) {
        o[i]     = (__bf16)(f0[i] * sc);
        o[4 + i] = (__bf16)(f1[i] * sc);
      }
      *(v8bf*)&sA[slot * 16 * LDS_STRIDE + row * LDS_STRIDE + col] = o;
      ++slot;
    }
    if constexpr (NACC >= 2) {
      float sc = 1.0f / fmaxf(c1[row0 + row], 1.0f);
      const float* p = A1 + (long)(row0 + row) * H_N + col;
      v4f f0 = *(const v4f*)p;
      v4f f1 = *(const v4f*)(p + 4);
      v8bf o;
#pragma unroll
      for (int i = 0; i < 4; ++i) {
        o[i]     = (__bf16)(f0[i] * sc);
        o[4 + i] = (__bf16)(f1[i] * sc);
      }
      *(v8bf*)&sA[slot * 16 * LDS_STRIDE + row * LDS_STRIDE + col] = o;
      ++slot;
    }
    if constexpr (HAS_X) {
      v8bf o = *(const v8bf*)(Xd + (long)(row0 + row) * H_N + col);
      *(v8bf*)&sA[slot * 16 * LDS_STRIDE + row * LDS_STRIDE + col] = o;
    }
  }
  __syncthreads();

  // ---- compute: LDS A fragments x pre-packed global B fragments ----
  v8f acc = {};
#pragma unroll
  for (int kt = 0; kt < 4; ++kt) {
    const int k0 = kt * 32;
    int slot = 0;
    if constexpr (NACC >= 1) {
      v16bf a = load_a_lds(&sA[slot * 16 * LDS_STRIDE], k0, lane);
      v16bf b = load_b_packed(W0p, kt, nt, ntiles, lane);
      acc = __builtin_amdgcn_wmma_f32_16x16x32_bf16(false, a, false, b,
                                                    (short)0, acc, false, false);
      ++slot;
    }
    if constexpr (NACC >= 2) {
      v16bf a = load_a_lds(&sA[slot * 16 * LDS_STRIDE], k0, lane);
      v16bf b = load_b_packed(W1p, kt, nt, ntiles, lane);
      acc = __builtin_amdgcn_wmma_f32_16x16x32_bf16(false, a, false, b,
                                                    (short)0, acc, false, false);
      ++slot;
    }
    if constexpr (HAS_X) {
      v16bf a = load_a_lds(&sA[slot * 16 * LDS_STRIDE], k0, lane);
      v16bf b = load_b_packed(Wxp, kt, nt, ntiles, lane);
      acc = __builtin_amdgcn_wmma_f32_16x16x32_bf16(false, a, false, b,
                                                    (short)0, acc, false, false);
    }
  }

  // epilogue: bias + leaky relu + store (C/D layout: col lane&15, rows v+8*half)
  int n = nt * 16 + (lane & 15);
  float bv = bias_a ? bias_a[n] : 0.0f;
  if (bias_b) bv += bias_b[n];
#pragma unroll
  for (int v = 0; v < 8; ++v) {
    int m = row0 + v + ((lane >> 4) << 3);
    float o = acc[v] + bv;
    if constexpr (RELU) o = (o > 0.0f) ? o : 0.01f * o;
    if constexpr (OUT_BF16)
      ((__bf16*)out)[(long)m * N + n] = (__bf16)o;
    else
      ((float*)out)[(long)m * N + n] = o;
  }
}

// ---------------------------------------------------------------------------

extern "C" void kernel_launch(void* const* d_in, const int* in_sizes, int n_in,
                              void* d_out, int out_size, void* d_ws, size_t ws_size,
                              hipStream_t stream) {
  const float* x_author = (const float*)d_in[0];   // [NA,128]
  const float* x_paper  = (const float*)d_in[1];   // [NP,128]
  const float* Wl       = (const float*)d_in[2];   // [L,3,128,128]
  const float* bl       = (const float*)d_in[3];   // [L,3,128]
  const float* Wr       = (const float*)d_in[4];   // [L,3,128,128]
  const float* W_out    = (const float*)d_in[5];   // [128,64]
  const float* b_out    = (const float*)d_in[6];   // [64]
  const int* writes_src = (const int*)d_in[7];
  const int* writes_dst = (const int*)d_in[8];
  const int* wb_src     = (const int*)d_in[9];
  const int* wb_dst     = (const int*)d_in[10];
  const int* cites_src  = (const int*)d_in[11];
  const int* cites_dst  = (const int*)d_in[12];
  (void)in_sizes; (void)n_in; (void)out_size; (void)ws_size;

  // ---- carve workspace ----
  char* ws = (char*)d_ws;
  auto carve = [&](size_t bytes) -> char* {
    char* p = ws;
    ws += (bytes + 255) & ~(size_t)255;
    return p;
  };
  __bf16* xa_bf[2], *xp_bf[2];
  xa_bf[0] = (__bf16*)carve((size_t)NA_N * H_N * 2);
  xa_bf[1] = (__bf16*)carve((size_t)NA_N * H_N * 2);
  xp_bf[0] = (__bf16*)carve((size_t)NP_N * H_N * 2);
  xp_bf[1] = (__bf16*)carve((size_t)NP_N * H_N * 2);
  float* acc_w  = (float*)carve((size_t)NP_N * H_N * 4);
  float* acc_c  = (float*)carve((size_t)NP_N * H_N * 4);
  float* acc_wb = (float*)carve((size_t)NA_N * H_N * 4);
  float* cnt_w  = (float*)carve((size_t)NP_N * 4);
  float* cnt_c  = (float*)carve((size_t)NP_N * 4);
  float* cnt_wb = (float*)carve((size_t)NA_N * 4);
  const int WB = H_N * H_N;                 // 16384 elems per weight matrix
  __bf16* wl_w[L_N], *wl_c[L_N], *wr_p[L_N], *wl_wb[L_N], *wr_a[L_N];
  for (int l = 0; l < L_N; ++l) {
    wl_w[l]  = (__bf16*)carve((size_t)WB * 2);
    wl_c[l]  = (__bf16*)carve((size_t)WB * 2);
    wr_p[l]  = (__bf16*)carve((size_t)WB * 2);
    wl_wb[l] = (__bf16*)carve((size_t)WB * 2);
    wr_a[l]  = (__bf16*)carve((size_t)WB * 2);
  }
  __bf16* wout_bf = (__bf16*)carve((size_t)H_N * OUT_N * 2);

  // ---- prep: cast activations, pack weights into WMMA B-fragment order ----
  {
    long n = (long)NA_N * H_N;
    cast_bf16_kernel<<<(unsigned)((n + 255) / 256), 256, 0, stream>>>(x_author, xa_bf[0], n);
    n = (long)NP_N * H_N;
    cast_bf16_kernel<<<(unsigned)((n + 255) / 256), 256, 0, stream>>>(x_paper, xp_bf[0], n);
  }
  for (int l = 0; l < L_N; ++l) {
    const float* Wl_l = Wl + (size_t)l * 3 * WB;
    const float* Wr_l = Wr + (size_t)l * 3 * WB;
    int g = (WB + 255) / 256;
    pack_b_f32_kernel<<<g, 256, 0, stream>>>(Wl_l + 0 * WB, nullptr,        wl_w[l],  H_N);
    pack_b_f32_kernel<<<g, 256, 0, stream>>>(Wl_l + 2 * WB, nullptr,        wl_c[l],  H_N);
    pack_b_f32_kernel<<<g, 256, 0, stream>>>(Wr_l + 0 * WB, Wr_l + 2 * WB,  wr_p[l],  H_N); // Wr0+Wr2
    pack_b_f32_kernel<<<g, 256, 0, stream>>>(Wl_l + 1 * WB, nullptr,        wl_wb[l], H_N);
    pack_b_f32_kernel<<<g, 256, 0, stream>>>(Wr_l + 1 * WB, nullptr,        wr_a[l],  H_N);
  }
  {
    int n = H_N * OUT_N;
    pack_b_f32_kernel<<<(n + 255) / 256, 256, 0, stream>>>(W_out, nullptr, wout_bf, OUT_N);
  }

  // ---- layers ----
  const int SC_BLOCKS = 4096, SC_THREADS = 256;   // 32768 waves, grid-stride over E
  int cur = 0;
  for (int l = 0; l < L_N; ++l) {
    int nxt = cur ^ 1;
    hipMemsetAsync(acc_w,  0, (size_t)NP_N * H_N * 4, stream);
    hipMemsetAsync(acc_c,  0, (size_t)NP_N * H_N * 4, stream);
    hipMemsetAsync(acc_wb, 0, (size_t)NA_N * H_N * 4, stream);
    hipMemsetAsync(cnt_w,  0, (size_t)NP_N * 4, stream);
    hipMemsetAsync(cnt_c,  0, (size_t)NP_N * 4, stream);
    hipMemsetAsync(cnt_wb, 0, (size_t)NA_N * 4, stream);

    // relation scatters (mean numerators + degrees)
    scatter_accum_kernel<<<SC_BLOCKS, SC_THREADS, 0, stream>>>(
        xa_bf[cur], writes_src, writes_dst, acc_w, cnt_w, E_N);       // author->paper
    scatter_accum_kernel<<<SC_BLOCKS, SC_THREADS, 0, stream>>>(
        xp_bf[cur], cites_src, cites_dst, acc_c, cnt_c, E_N);         // paper->paper
    scatter_accum_kernel<<<SC_BLOCKS, SC_THREADS, 0, stream>>>(
        xp_bf[cur], wb_src, wb_dst, acc_wb, cnt_wb, E_N);             // paper->author

    const float* bl_l = bl + (size_t)l * 3 * H_N;
    // papers: mean_w@Wl0 + mean_c@Wl2 + xp@(Wr0+Wr2) + b0 + b2, leaky relu
    sage_gemm_kernel<2, true, true, true><<<NP_N / 16, 256, 0, stream>>>(
        acc_w, cnt_w, acc_c, cnt_c, xp_bf[cur],
        wl_w[l], wl_c[l], wr_p[l], bl_l + 0 * H_N, bl_l + 2 * H_N,
        xp_bf[nxt], H_N);
    // authors: mean_wb@Wl1 + xa@Wr1 + b1, leaky relu
    sage_gemm_kernel<1, true, true, true><<<NA_N / 16, 256, 0, stream>>>(
        acc_wb, cnt_wb, nullptr, nullptr, xa_bf[cur],
        wl_wb[l], nullptr, wr_a[l], bl_l + 1 * H_N, nullptr,
        xa_bf[nxt], H_N);
    cur = nxt;
  }

  // ---- final projection: xa @ W_out + b_out -> f32 d_out [NA, 64] ----
  sage_gemm_kernel<0, true, false, false><<<NA_N / 16, 128, 0, stream>>>(
      nullptr, nullptr, nullptr, nullptr, xa_bf[cur],
      nullptr, nullptr, wout_bf, b_out, nullptr,
      d_out, OUT_N);
}
